// SelfMatchingLayer_78847009620109
// MI455X (gfx1250) — matrix operand order repaired and checked by reading, not compiled, not run on hardware
//
#include <hip/hip_runtime.h>
#include <hip/hip_bf16.h>

typedef __attribute__((ext_vector_type(16))) _Float16 v16h;
typedef __attribute__((ext_vector_type(8)))  _Float16 v8h;
typedef __attribute__((ext_vector_type(8)))  float    v8f;

#define BDIM 32
#define LDIM 512
#define HDIM 256
#define G3H  768           // 3*H
#define BL   (BDIM*LDIM)   // 16384
#define TT   4             // timesteps per attention block

// ---------------------------------------------------------------------------
// Fast transcendentals: gfx1250 has a hardware v_tanh_f32 (TRANS32 unit)
// ---------------------------------------------------------------------------
__device__ __forceinline__ float fast_tanh(float x) {
#if __has_builtin(__builtin_amdgcn_tanhf)
    return __builtin_amdgcn_tanhf(x);
#elif __has_builtin(__builtin_amdgcn_tanh_f32)
    return __builtin_amdgcn_tanh_f32(x);
#else
    float y;
    // v_nop satisfies the 1-op TRANS co-execution hazard rule (ISA 7.4)
    asm volatile("v_tanh_f32 %0, %1\n\tv_nop" : "=v"(y) : "v"(x));
    return y;
#endif
}
// sigmoid(x) = 0.5*tanh(x/2) + 0.5 : one TRANS op instead of exp+rcp
__device__ __forceinline__ float fast_sigmoid(float x) {
    return fmaf(0.5f, fast_tanh(0.5f * x), 0.5f);
}

// ---------------------------------------------------------------------------
// WMMA fragment helpers (wave32 layouts per CDNA5 ISA 7.12.2)
// ---------------------------------------------------------------------------
__device__ __forceinline__ v16h mk16(v8h lo, v8h hi) {
    v16h a;
#pragma unroll
    for (int i = 0; i < 8; ++i) { a[i] = lo[i]; a[i + 8] = hi[i]; }
    return a;
}

// A-matrix 16x32 f16: lane holds row m=lane&15; halves 0..7 -> K=8g+e,
// halves 8..15 -> K=16+8g+(e-8). Two 16B loads per lane.
__device__ __forceinline__ v16h load_a_frag(const _Float16* __restrict__ X,
                                            int row, int ld, int kb, int lane) {
    int g = lane >> 4;
    const _Float16* p = X + (size_t)row * ld + kb + g * 8;
    v8h lo = *(const v8h*)p;
    v8h hi = *(const v8h*)(p + 16);
    return mk16(lo, hi);
}

// B-matrix 32x16 f16 for out = X @ W^T: column n of B == row n of W (N,K)
// row-major, so K is contiguous: halves 0..7 -> K=8g+e, 8..15 -> K=16+8g+(e-8). Two 16B loads.
__device__ __forceinline__ v16h load_b_frag(const _Float16* __restrict__ W,
                                            int col, int ld, int kb, int lane) {
    int g = lane >> 4;
    const _Float16* p = W + (size_t)col * ld + kb + g * 8;
    v8h lo = *(const v8h*)p;
    v8h hi = *(const v8h*)(p + 16);
    return mk16(lo, hi);
}

// ---------------------------------------------------------------------------
// Generic batched GEMM: out(MxN,f32) = act(X16(MxK) @ W16(N,K)^T + bias)
// block = 4 waves, one 16x16 tile per wave. act: 0=none, 1=sigmoid.
// ---------------------------------------------------------------------------
__global__ __launch_bounds__(128)
void wmma_gemm_xwT(const _Float16* __restrict__ X, const _Float16* __restrict__ W,
                   const float* __restrict__ bias, float* __restrict__ out,
                   int M, int N, int K, int act) {
    int lane  = threadIdx.x & 31;
    int wave  = threadIdx.x >> 5;
    int nTile = blockIdx.x;
    int mTile = blockIdx.y * 4 + wave;
    if (mTile * 16 >= M) return;                 // wave-uniform

    v8f acc = {};
    int arow = mTile * 16 + (lane & 15);
    int bcol = nTile * 16 + (lane & 15);
    for (int kb = 0; kb < K; kb += 32) {
        v16h a = load_a_frag(X, arow, K, kb, lane);
        v16h b = load_b_frag(W, bcol, K, kb, lane);
        acc = __builtin_amdgcn_wmma_f32_16x16x32_f16(false, a, false, b,
                                                     (short)0, acc, false, false);
    }
    int g = lane >> 4;
    int n = nTile * 16 + (lane & 15);
    float bv = bias ? bias[n] : 0.0f;
#pragma unroll
    for (int r = 0; r < 8; ++r) {
        int m = mTile * 16 + r + 8 * g;
        float v = acc[r] + bv;
        if (act == 1) v = fast_sigmoid(v);
        out[(size_t)m * N + n] = v;
    }
}

// ---------------------------------------------------------------------------
// Elementwise converters
// ---------------------------------------------------------------------------
__global__ void f32_to_f16_kernel(const float* __restrict__ in,
                                  _Float16* __restrict__ out, size_t n) {
    for (size_t i = (size_t)blockIdx.x * blockDim.x + threadIdx.x; i < n;
         i += (size_t)gridDim.x * blockDim.x)
        out[i] = (_Float16)in[i];
}

// P -> P16 and left half of the concat buffer XC16 (rows of width 2H)
__global__ void pack_P_kernel(const float* __restrict__ P,
                              _Float16* __restrict__ P16,
                              _Float16* __restrict__ XC16) {
    size_t i = (size_t)blockIdx.x * blockDim.x + threadIdx.x;  // over B*L*H
    if (i >= (size_t)BL * HDIM) return;
    size_t row = i / HDIM, col = i % HDIM;
    _Float16 v = (_Float16)P[i];
    P16[i] = v;
    XC16[row * (2 * HDIM) + col] = v;
}

// c_gated = g * c  (f16 for the GRU input GEMMs)
__global__ void gate_mul_kernel(const float* __restrict__ G,
                                const float* __restrict__ C,
                                _Float16* __restrict__ CG16, size_t n) {
    for (size_t i = (size_t)blockIdx.x * blockDim.x + threadIdx.x; i < n;
         i += (size_t)gridDim.x * blockDim.x)
        CG16[i] = (_Float16)(G[i] * C[i]);
}

// ---------------------------------------------------------------------------
// Fused attention: one block per (b, 4 consecutive t). scores -> softmax ->
// context. Hq[b] / P[b] are read once per 4 timesteps (4x L2 traffic cut).
// Writes C32[b,t,:] and the right half of XC16.
// ---------------------------------------------------------------------------
__global__ __launch_bounds__(256)
void attention_kernel(const float* __restrict__ P, const float* __restrict__ Hq,
                      const float* __restrict__ Pp, const float* __restrict__ ws_w,
                      const float* __restrict__ ws_b,
                      float* __restrict__ C32, _Float16* __restrict__ XC16) {
    int b  = blockIdx.x >> 7;          // / (512/TT)
    int t0 = (blockIdx.x & 127) * TT;
    __shared__ float ppt[TT][HDIM];
    __shared__ float sc[TT][LDIM];
    __shared__ float red[64];
    int tid  = threadIdx.x;
    int lane = tid & 31;
    int wave = tid >> 5;               // 8 waves

#pragma unroll
    for (int tt = 0; tt < TT; ++tt)
        ppt[tt][tid] = Pp[((size_t)b * LDIM + t0 + tt) * HDIM + tid];
    __syncthreads();

    // hoisted per-lane constants
    float wv[8], pv[TT][8];
#pragma unroll
    for (int jj = 0; jj < 8; ++jj) {
        int h = lane + 32 * jj;
        wv[jj] = ws_w[h];
#pragma unroll
        for (int tt = 0; tt < TT; ++tt) pv[tt][jj] = ppt[tt][h];
    }
    float wsb = ws_b[0];

    // phase 1: scores[tt][l] = ws . tanh(Hq[b,l,:] + Pp[b,t0+tt,:]) + ws_b
    for (int l = wave; l < LDIM; l += 8) {
        const float* hq = Hq + ((size_t)b * LDIM + l) * HDIM;
        float hv[8];
#pragma unroll
        for (int jj = 0; jj < 8; ++jj) hv[jj] = hq[lane + 32 * jj];
        float s[TT] = {};
#pragma unroll
        for (int jj = 0; jj < 8; ++jj)
#pragma unroll
            for (int tt = 0; tt < TT; ++tt)
                s[tt] += fast_tanh(hv[jj] + pv[tt][jj]) * wv[jj];
#pragma unroll
        for (int tt = 0; tt < TT; ++tt) {
            float v = s[tt];
            for (int off = 16; off; off >>= 1) v += __shfl_down(v, off, 32);
            if (lane == 0) sc[tt][l] = v + wsb;
        }
    }
    __syncthreads();

    // phase 2: softmax per tt (max / exp / sum); normalization folded later
    for (int tt = 0; tt < TT; ++tt) {
        float m2 = fmaxf(sc[tt][tid], sc[tt][tid + 256]);
        for (int off = 16; off; off >>= 1) m2 = fmaxf(m2, __shfl_down(m2, off, 32));
        if (lane == 0) red[wave] = m2;
        __syncthreads();
        if (tid == 0) {
            float mm = red[0];
            for (int i = 1; i < 8; ++i) mm = fmaxf(mm, red[i]);
            red[8] = mm;
        }
        __syncthreads();
        float mx = red[8];
        float e0 = __expf(sc[tt][tid] - mx);
        float e1 = __expf(sc[tt][tid + 256] - mx);
        sc[tt][tid] = e0; sc[tt][tid + 256] = e1;
        float ss = e0 + e1;
        for (int off = 16; off; off >>= 1) ss += __shfl_down(ss, off, 32);
        if (lane == 0) red[16 + wave] = ss;
        __syncthreads();
        if (tid == 0) {
            float tsum = red[16];
            for (int i = 1; i < 8; ++i) tsum += red[16 + i];
            red[48 + tt] = 1.0f / tsum;
        }
        __syncthreads();
    }
    float inv[TT];
#pragma unroll
    for (int tt = 0; tt < TT; ++tt) inv[tt] = red[48 + tt];

    // phase 3: c[tt][h] = inv * sum_l e[tt][l] * P[b,l,h]   (thread h)
    float acc[TT] = {};
    const float* pb = P + (size_t)b * LDIM * HDIM + tid;
    for (int l = 0; l < LDIM; ++l) {
        float p = pb[(size_t)l * HDIM];
#pragma unroll
        for (int tt = 0; tt < TT; ++tt) acc[tt] += sc[tt][l] * p;
    }
#pragma unroll
    for (int tt = 0; tt < TT; ++tt) {
        float v = acc[tt] * inv[tt];
        size_t row = (size_t)b * LDIM + t0 + tt;
        C32[row * HDIM + tid] = v;
        XC16[row * (2 * HDIM) + HDIM + tid] = (_Float16)v;
    }
}

// ---------------------------------------------------------------------------
// Persistent GRU scan: one workgroup, 32 waves. h kept in LDS (f16 for WMMA
// A-fragments with padded stride, f32 for the update). Per step: gh = h@Whh^T
// via 24 WMMAs (each wave owns one (m,n) patch of ALL three gates, so the
// gate mix is done straight on the accumulators), then elementwise GRU.
// Next-step gi rows are prefetched (global_prefetch) under the GEMM.
// ---------------------------------------------------------------------------
#define HS16 264   // padded f16 row stride (bank-conflict avoidance)
#define HS32 257

__global__ __launch_bounds__(1024)
void gru_scan_kernel(const float* __restrict__ GI,      // (B*L, 768), bih added
                     const _Float16* __restrict__ Whh,  // (768, 256) f16
                     const float* __restrict__ bhh,     // (768,)
                     float* __restrict__ out,           // (B, L, 2H)
                     int outOff, int reverse) {
    __shared__ _Float16 h16[BDIM * HS16];
    __shared__ float    h32[BDIM * HS32];
    int tid  = threadIdx.x;
    int lane = tid & 31;
    int wave = tid >> 5;          // 0..31
    int mTile = wave >> 4;        // 0..1   (batch rows 0-15 / 16-31)
    int nTile = wave & 15;        // 0..15  (hidden cols, 16 per tile)

    for (int i = tid; i < BDIM * HS16; i += 1024) h16[i] = (_Float16)0;
    for (int i = tid; i < BDIM * HS32; i += 1024) h32[i] = 0.0f;
    __syncthreads();

    int g    = lane >> 4;
    int ln   = lane & 15;
    int arow = mTile * 16 + ln;   // batch row for A-fragment
    int j    = nTile * 16 + ln;   // hidden index for B-column / elementwise
    float bR = bhh[j], bZ = bhh[HDIM + j], bN = bhh[2 * HDIM + j];

    for (int s = 0; s < LDIM; ++s) {
        int t = reverse ? (LDIM - 1 - s) : s;
        // prefetch next step's gi rows while the GEMM below runs
        if (s + 1 < LDIM) {
            int tn = reverse ? (LDIM - 2 - s) : (s + 1);
#pragma unroll
            for (int r = 0; r < 8; ++r) {
                int bb = mTile * 16 + r + 8 * g;
                const float* pp = &GI[((size_t)bb * LDIM + tn) * G3H + j];
                __builtin_prefetch(pp, 0, 1);
                __builtin_prefetch(pp + HDIM, 0, 1);
                __builtin_prefetch(pp + 2 * HDIM, 0, 1);
            }
        }
        v8f aR = {}; v8f aZ = {}; v8f aN = {};
#pragma unroll
        for (int kb = 0; kb < HDIM; kb += 32) {
            const _Float16* pa = &h16[arow * HS16 + kb + g * 8];
            v16h a = mk16(*(const v8h*)pa, *(const v8h*)(pa + 16));
            v16h b0 = load_b_frag(Whh, j,            HDIM, kb, lane);
            v16h b1 = load_b_frag(Whh, HDIM + j,     HDIM, kb, lane);
            v16h b2 = load_b_frag(Whh, 2 * HDIM + j, HDIM, kb, lane);
            aR = __builtin_amdgcn_wmma_f32_16x16x32_f16(false, a, false, b0,
                                                        (short)0, aR, false, false);
            aZ = __builtin_amdgcn_wmma_f32_16x16x32_f16(false, a, false, b1,
                                                        (short)0, aZ, false, false);
            aN = __builtin_amdgcn_wmma_f32_16x16x32_f16(false, a, false, b2,
                                                        (short)0, aN, false, false);
        }
        float hnew[8];
#pragma unroll
        for (int r = 0; r < 8; ++r) {
            int bb = mTile * 16 + r + 8 * g;  // batch index (D-layout row)
            size_t gbase = ((size_t)bb * LDIM + t) * G3H;
            float ir  = GI[gbase + j];
            float iz  = GI[gbase + HDIM + j];
            float inn = GI[gbase + 2 * HDIM + j];
            float rr = fast_sigmoid(ir + aR[r] + bR);
            float zz = fast_sigmoid(iz + aZ[r] + bZ);
            float nn = fast_tanh(inn + rr * (aN[r] + bN));
            float ho = h32[bb * HS32 + j];
            hnew[r] = (1.0f - zz) * nn + zz * ho;
        }
        __syncthreads();   // all waves done reading h before updating it
#pragma unroll
        for (int r = 0; r < 8; ++r) {
            int bb = mTile * 16 + r + 8 * g;
            h32[bb * HS32 + j] = hnew[r];
            h16[bb * HS16 + j] = (_Float16)hnew[r];
            out[((size_t)bb * LDIM + t) * (2 * HDIM) + outOff + j] = hnew[r];
        }
        __syncthreads();
    }
}

// ---------------------------------------------------------------------------
extern "C" void kernel_launch(void* const* d_in, const int* in_sizes, int n_in,
                              void* d_out, int out_size, void* d_ws, size_t ws_size,
                              hipStream_t stream) {
    const float* P     = (const float*)d_in[0];
    const float* wq_w  = (const float*)d_in[1];
    const float* wq_b  = (const float*)d_in[2];
    const float* wp_w  = (const float*)d_in[3];
    const float* wp_b  = (const float*)d_in[4];
    const float* ws_w  = (const float*)d_in[5];
    const float* ws_b  = (const float*)d_in[6];
    const float* wg_w  = (const float*)d_in[7];
    const float* wg_b  = (const float*)d_in[8];
    const float* Wih_l = (const float*)d_in[9];
    const float* Whh_l = (const float*)d_in[10];
    const float* bih_l = (const float*)d_in[11];
    const float* bhh_l = (const float*)d_in[12];
    const float* Wih_r = (const float*)d_in[13];
    const float* Whh_r = (const float*)d_in[14];
    const float* bih_r = (const float*)d_in[15];
    const float* bhh_r = (const float*)d_in[16];
    float* out = (float*)d_out;

    // workspace carve-up
    char* ws = (char*)d_ws;
    size_t off = 0;
    auto alloc = [&](size_t bytes) -> char* {
        char* p = ws + off;
        off += (bytes + 255) & ~(size_t)255;
        return p;
    };
    _Float16* P16    = (_Float16*)alloc((size_t)BL * HDIM * 2);
    _Float16* XC16   = (_Float16*)alloc((size_t)BL * 2 * HDIM * 2);
    _Float16* wq16   = (_Float16*)alloc((size_t)HDIM * HDIM * 2);
    _Float16* wp16   = (_Float16*)alloc((size_t)HDIM * HDIM * 2);
    _Float16* wg16   = (_Float16*)alloc((size_t)HDIM * 2 * HDIM * 2);
    _Float16* WihL16 = (_Float16*)alloc((size_t)G3H * HDIM * 2);
    _Float16* WhhL16 = (_Float16*)alloc((size_t)G3H * HDIM * 2);
    _Float16* WihR16 = (_Float16*)alloc((size_t)G3H * HDIM * 2);
    _Float16* WhhR16 = (_Float16*)alloc((size_t)G3H * HDIM * 2);
    float*    Hq32   = (float*)alloc((size_t)BL * HDIM * 4);
    float*    Pp32   = (float*)alloc((size_t)BL * HDIM * 4);
    float*    C32    = (float*)alloc((size_t)BL * HDIM * 4);
    float*    G32    = (float*)alloc((size_t)BL * HDIM * 4);
    _Float16* CG16   = (_Float16*)alloc((size_t)BL * HDIM * 2);
    float*    GIL    = (float*)alloc((size_t)BL * G3H * 4);
    float*    GIR    = (float*)alloc((size_t)BL * G3H * 4);
    (void)ws_size; (void)in_sizes; (void)n_in; (void)out_size;

    // 1. precision packing
    pack_P_kernel<<<(BL * HDIM + 255) / 256, 256, 0, stream>>>(P, P16, XC16);
    f32_to_f16_kernel<<<256, 256, 0, stream>>>(wq_w,  wq16,   (size_t)HDIM * HDIM);
    f32_to_f16_kernel<<<256, 256, 0, stream>>>(wp_w,  wp16,   (size_t)HDIM * HDIM);
    f32_to_f16_kernel<<<256, 256, 0, stream>>>(wg_w,  wg16,   (size_t)HDIM * 2 * HDIM);
    f32_to_f16_kernel<<<256, 256, 0, stream>>>(Wih_l, WihL16, (size_t)G3H * HDIM);
    f32_to_f16_kernel<<<256, 256, 0, stream>>>(Whh_l, WhhL16, (size_t)G3H * HDIM);
    f32_to_f16_kernel<<<256, 256, 0, stream>>>(Wih_r, WihR16, (size_t)G3H * HDIM);
    f32_to_f16_kernel<<<256, 256, 0, stream>>>(Whh_r, WhhR16, (size_t)G3H * HDIM);

    // 2. Hq = P@wq^T + b, Pp = P@wp^T + b      (WMMA, M=16384 K=256 N=256)
    dim3 gq(HDIM / 16, BL / 64);
    wmma_gemm_xwT<<<gq, 128, 0, stream>>>(P16, wq16, wq_b, Hq32, BL, HDIM, HDIM, 0);
    wmma_gemm_xwT<<<gq, 128, 0, stream>>>(P16, wp16, wp_b, Pp32, BL, HDIM, HDIM, 0);

    // 3. attention: scores/softmax/context, 4 timesteps per block
    attention_kernel<<<BL / TT, 256, 0, stream>>>(P, Hq32, Pp32, ws_w, ws_b, C32, XC16);

    // 4. gate g = sigmoid([p;c]@wg^T + b)      (WMMA, K=512, fused sigmoid)
    wmma_gemm_xwT<<<gq, 128, 0, stream>>>(XC16, wg16, wg_b, G32, BL, HDIM, 2 * HDIM, 1);

    // 5. c_gated = g*c (f16)
    gate_mul_kernel<<<1024, 256, 0, stream>>>(G32, C32, CG16, (size_t)BL * HDIM);

    // 6. batched GRU input gates: gi = c_gated@Wih^T + bih   (WMMA, N=768)
    dim3 gg(G3H / 16, BL / 64);
    wmma_gemm_xwT<<<gg, 128, 0, stream>>>(CG16, WihL16, bih_l, GIL, BL, G3H, HDIM, 0);
    wmma_gemm_xwT<<<gg, 128, 0, stream>>>(CG16, WihR16, bih_r, GIR, BL, G3H, HDIM, 0);

    // 7. serial scans: persistent single-workgroup WMMA GRU per direction
    gru_scan_kernel<<<1, 1024, 0, stream>>>(GIL, WhhL16, bhh_l, out, 0, 0);
    gru_scan_kernel<<<1, 1024, 0, stream>>>(GIR, WhhR16, bhh_r, out, HDIM, 1);
}
